// MultiHeadAttention_43971875177057
// MI455X (gfx1250) — matrix operand order repaired and checked by reading, not compiled, not run on hardware
//
#include <hip/hip_runtime.h>

typedef __attribute__((ext_vector_type(16))) _Float16 v16h;
typedef __attribute__((ext_vector_type(4)))  _Float16 v4h;
typedef __attribute__((ext_vector_type(8)))  float    v8f;
typedef __attribute__((ext_vector_type(4)))  float    v4f;
typedef __attribute__((ext_vector_type(4)))  int      v4i;

#define D_MODEL 512
#define SEQ     4096
#define NHEAD   8
#define DK      64
#define BATCH   2

// ---------------------------------------------------------------------------
// gfx1250 async global->LDS path (ASYNCcnt-tracked), signature probe-verified:
// (v4i addrspace(1)* src, v4i addrspace(3)* dst, imm int offset, imm int cpol)
// ---------------------------------------------------------------------------
#if defined(__has_builtin)
#if __has_builtin(__builtin_amdgcn_global_load_async_to_lds_b128) && \
    __has_builtin(__builtin_amdgcn_s_wait_asynccnt)
#define USE_ASYNC_LDS 1
#endif
#endif
#ifndef USE_ASYNC_LDS
#define USE_ASYNC_LDS 0
#endif

#if USE_ASYNC_LDS
#pragma message("CDNA5: async global->LDS path ENABLED")
typedef __attribute__((address_space(1))) v4i* gbl_i4p;
typedef __attribute__((address_space(3))) v4i* lds_i4p;
#else
#pragma message("CDNA5: async global->LDS path DISABLED (fallback gathers)")
#endif

// ---------------------------------------------------------------------------
// WMMA helpers (CDNA5: D = A(16x32 f16) * B(32x16 f16) + C(16x16 f32))
// ---------------------------------------------------------------------------
__device__ __forceinline__ v8f wmma16(v16h a, v16h b, v8f c) {
  return __builtin_amdgcn_wmma_f32_16x16x32_f16(
      false, a, false, b, (short)0, c, false, false);
}

// A-matrix 16x32 f16: lane L(0-15) row M=L holds K=0..15;
// lane L+16 holds K=16..31.  Source: f16 row, two 16B runs.
__device__ __forceinline__ v16h loadA_h(const _Float16* __restrict__ row,
                                        int k0, int kh) {
  v16h d;
  const _Float16* p1 = row + k0 + kh * 16;
  const _Float16* p2 = row + k0 + kh * 16 + 8;
#pragma unroll
  for (int i = 0; i < 8; ++i) d[i]     = p1[i];
#pragma unroll
  for (int i = 0; i < 8; ++i) d[i + 8] = p2[i];
  return d;
}

// B-matrix 32x16 f16: lane L(0-15) col N=L holds K=0..15; lane L+16 K=16..31.
// Source: 16 contiguous halfs (two b128 loads).
__device__ __forceinline__ v16h loadB16_h(const _Float16* __restrict__ p) {
  v16h d;
#pragma unroll
  for (int i = 0; i < 16; ++i) d[i] = p[i];
  return d;
}

// Half-wave (16-lane) reductions matching the C-matrix row layout.
__device__ __forceinline__ float rowmax16(float v) {
#pragma unroll
  for (int off = 1; off < 16; off <<= 1) v = fmaxf(v, __shfl_xor(v, off, 16));
  return v;
}
__device__ __forceinline__ float rowsum16(float v) {
#pragma unroll
  for (int off = 1; off < 16; off <<= 1) v += __shfl_xor(v, off, 16);
  return v;
}

// ---------------------------------------------------------------------------
// Kernel 0: f32 -> f16 conversion (x and the four weight matrices).
// ---------------------------------------------------------------------------
__global__ void cvt_f32_f16(const float* __restrict__ src,
                            _Float16* __restrict__ dst, int n4) {
  const int i = blockIdx.x * blockDim.x + threadIdx.x;
  if (i < n4) {
    v4f a = ((const v4f*)src)[i];
    v4h d;
#pragma unroll
    for (int j = 0; j < 4; ++j) d[j] = (_Float16)a[j];
    ((v4h*)dst)[i] = d;
  }
}

// ---------------------------------------------------------------------------
// Kernel 1: fused Q/K/V projection (f16 in, f16 out, f32 accumulate).
// Q -> [b,h,s,dk] * 1/sqrt(dk);  K -> [b,h,s,dk];  V -> [b,h,dk,s] (transposed
// so attention's P*V B-operand is row-contiguous).
// ---------------------------------------------------------------------------
__global__ void qkv_kernel(const _Float16* __restrict__ xh,
                           const _Float16* __restrict__ Whq,
                           const _Float16* __restrict__ Whk,
                           const _Float16* __restrict__ Whv,
                           const float* __restrict__ bq,
                           const float* __restrict__ bk,
                           const float* __restrict__ bv,
                           _Float16* __restrict__ Qh,
                           _Float16* __restrict__ Kh,
                           _Float16* __restrict__ VhT) {
  const int l    = threadIdx.x;
  const int lane = l & 15, kh = l >> 4;
  const int n0    = blockIdx.x * 16;
  const int m0    = blockIdx.y * 16;
  const int which = blockIdx.z;                 // 0:Q 1:K 2:V

  const _Float16* W  = (which == 0) ? Whq : (which == 1) ? Whk : Whv;
  const float* bias  = (which == 0) ? bq  : (which == 1) ? bk  : bv;
  _Float16*    Out   = (which == 0) ? Qh  : (which == 1) ? Kh  : VhT;

  const _Float16* xrow = xh + (size_t)(m0 + lane) * D_MODEL;
  const _Float16* wrow = W  + (size_t)(n0 + lane) * D_MODEL;

  v8f acc = {};
#pragma unroll
  for (int k0 = 0; k0 < D_MODEL; k0 += 32) {
    v16h a = loadA_h(xrow, k0, kh);
    v16h b = loadB16_h(wrow + k0 + kh * 16);    // B[k,n] = W[n,k]
    acc = wmma16(a, b, acc);
  }

  const float bn  = bias[n0 + lane];
  const float osc = (which == 0) ? 0.125f : 1.0f;   // fold 1/sqrt(dk) into Q
  const int   N   = n0 + lane;
  const int   hd  = N >> 6, dkk = N & (DK - 1);
#pragma unroll
  for (int r = 0; r < 8; ++r) {
    const int M = m0 + r + 8 * kh;
    const int b = M >> 12, s = M & (SEQ - 1);
    const float val = (acc[r] + bn) * osc;
    const size_t hb = (size_t)(b * NHEAD + hd);
    if (which == 2)
      Out[(hb * DK + dkk) * SEQ + s] = (_Float16)val;     // transposed V
    else
      Out[(hb * SEQ + s) * DK + dkk] = (_Float16)val;
  }
}

// ---------------------------------------------------------------------------
// Kernel 2: flash attention.  One wave per (batch, head, 16-query tile).
// K/V^T blocks double-buffered through LDS with async global->LDS: block i+1's
// 16 loads are issued, then s_wait_asynccnt(16) guarantees block i landed
// (async loads complete in order) while i+1's DMA overlaps compute.
// ---------------------------------------------------------------------------
#define KB_STRIDE 72   // halfs per kbuf row (144B, 16B aligned, bank-spread)
#define VB_STRIDE 40   // halfs per vbuf row (80B, 16B aligned)
#define P_STRIDE  40   // halfs per plds row (80B => aligned b128 gathers)

__global__ void attn_kernel(const _Float16* __restrict__ Qh,
                            const _Float16* __restrict__ Kh,
                            const _Float16* __restrict__ VhT,
                            const int*      __restrict__ mask,
                            _Float16* __restrict__ Ctxh) {
  __shared__ _Float16 plds[16 * P_STRIDE];      // 16x32 P tile (C->A relayout)
#if USE_ASYNC_LDS
  __shared__ _Float16 kbuf[2][32 * KB_STRIDE];  // K block: 32 keys x 64 halfs
  __shared__ _Float16 vbuf[2][64 * VB_STRIDE];  // V^T block: 64 dk x 32 halfs
#endif

  const int l    = threadIdx.x;
  const int lane = l & 15, kh = l >> 4;
  const int q0    = blockIdx.x * 16;
  const int head  = blockIdx.y;
  const int batch = blockIdx.z;

  const size_t hbase = (size_t)(batch * NHEAD + head) * SEQ * DK;
  const _Float16* Q  = Qh  + hbase;
  const _Float16* K  = Kh  + hbase;
  const _Float16* VT = VhT + hbase;             // [dk][s], row stride SEQ

  // Q tile (16x64, already scaled) as two A operands.
  const _Float16* qrow = Q + (size_t)(q0 + lane) * DK;
  const v16h qa0 = loadA_h(qrow, 0,  kh);
  const v16h qa1 = loadA_h(qrow, 32, kh);

  v8f acc[4];
#pragma unroll
  for (int t = 0; t < 4; ++t) acc[t] = {};
  float mrow[8], lrow[8];
#pragma unroll
  for (int r = 0; r < 8; ++r) { mrow[r] = -1e30f; lrow[r] = 0.f; }

#if USE_ASYNC_LDS
  // Stage one K block (4KB) + V^T block (4KB): 8+8 async b128 issues, 512B ea.
  auto stage = [&](int key0, int buf) {
#pragma unroll
    for (int j = 0; j < 8; ++j) {
      const int seg = j * 32 + l;
      const int kr = seg >> 3, ko = (seg & 7) * 8;     // K row / half-offset
      __builtin_amdgcn_global_load_async_to_lds_b128(
          (gbl_i4p)(K + (size_t)(key0 + kr) * DK + ko),
          (lds_i4p)(&kbuf[buf][kr * KB_STRIDE + ko]), 0, 0);
      const int vr = seg >> 2, vo = (seg & 3) * 8;     // V^T row / half-offset
      __builtin_amdgcn_global_load_async_to_lds_b128(
          (gbl_i4p)(VT + (size_t)vr * SEQ + key0 + vo),
          (lds_i4p)(&vbuf[buf][vr * VB_STRIDE + vo]), 0, 0);
    }
  };
  stage(0, 0);
#endif

  for (int key0 = 0; key0 < SEQ; key0 += 32) {
    const int cur = (key0 >> 5) & 1;

#if USE_ASYNC_LDS
    if (key0 + 32 < SEQ) {
      stage(key0 + 32, cur ^ 1);                 // overlap next block's DMA
      __builtin_amdgcn_s_wait_asynccnt(16);      // block `cur` has landed
    } else {
      __builtin_amdgcn_s_wait_asynccnt(0);
    }
    const _Float16* kb = kbuf[cur];
    const _Float16* vbl = vbuf[cur];
#else
    __builtin_prefetch(K + (size_t)(key0 + 32 + lane) * DK, 0, 1);
#endif

    // ---- scores: two 16x16 tiles (keys key0..+15 and key0+16..+31)
    v8f s0 = {}, s1 = {};
#if USE_ASYNC_LDS
    s0 = wmma16(qa0, loadB16_h(kb + lane * KB_STRIDE + kh * 16), s0);
    s0 = wmma16(qa1, loadB16_h(kb + lane * KB_STRIDE + 32 + kh * 16), s0);
    s1 = wmma16(qa0, loadB16_h(kb + (16 + lane) * KB_STRIDE + kh * 16), s1);
    s1 = wmma16(qa1, loadB16_h(kb + (16 + lane) * KB_STRIDE + 32 + kh * 16), s1);
#else
    const _Float16* krow0 = K + (size_t)(key0 + lane) * DK;
    const _Float16* krow1 = K + (size_t)(key0 + 16 + lane) * DK;
    s0 = wmma16(qa0, loadB16_h(krow0 + kh * 16), s0);
    s0 = wmma16(qa1, loadB16_h(krow0 + 32 + kh * 16), s0);
    s1 = wmma16(qa0, loadB16_h(krow1 + kh * 16), s1);
    s1 = wmma16(qa1, loadB16_h(krow1 + 32 + kh * 16), s1);
#endif

    // ---- mask + online softmax (row m = r + 8*kh lives in a 16-lane half)
    float p0[8], p1[8], scl[8];
#pragma unroll
    for (int r = 0; r < 8; ++r) {
      const int qi = q0 + r + 8 * kh;
      const int* mp = mask + (size_t)qi * SEQ + key0 + lane;
      float v0 = (mp[0]  != 0) ? -1e9f : s0[r];
      float v1 = (mp[16] != 0) ? -1e9f : s1[r];
      float mx = rowmax16(fmaxf(v0, v1));
      float mn = fmaxf(mrow[r], mx);
      float sc = __expf(mrow[r] - mn);
      float e0 = __expf(v0 - mn);
      float e1 = __expf(v1 - mn);
      lrow[r] = lrow[r] * sc + rowsum16(e0 + e1);
      mrow[r] = mn;
      scl[r]  = sc;
      p0[r] = e0; p1[r] = e1;
    }
#pragma unroll
    for (int t = 0; t < 4; ++t)
#pragma unroll
      for (int r = 0; r < 8; ++r) acc[t][r] *= scl[r];

    // ---- re-layout P (C-format 16x32 f32) -> A-format f16 via LDS
    __syncthreads();
#pragma unroll
    for (int r = 0; r < 8; ++r) {
      const int row = r + 8 * kh;
      plds[row * P_STRIDE + lane]      = (_Float16)p0[r];
      plds[row * P_STRIDE + 16 + lane] = (_Float16)p1[r];
    }
    __syncthreads();
    v16h pa;
#pragma unroll
    for (int e = 0; e < 8; ++e) pa[e]     = plds[lane * P_STRIDE + kh * 16 + e];
#pragma unroll
    for (int e = 0; e < 8; ++e) pa[e + 8] = plds[lane * P_STRIDE + kh * 16 + 8 + e];

    // ---- P (16x32) x V-block (32x64): four 16x16 output tiles
#pragma unroll
    for (int t = 0; t < 4; ++t) {
#if USE_ASYNC_LDS
      v16h vb = loadB16_h(vbl + (t * 16 + lane) * VB_STRIDE + kh * 16);
#else
      v16h vb = loadB16_h(VT + (size_t)(t * 16 + lane) * SEQ + key0 + kh * 16);
#endif
      acc[t] = wmma16(pa, vb, acc[t]);          // B[kk,n] = V^T[n0+n, key0+kk]
    }
  }

  // ---- normalize, store ctx as f16 in [b,h,s,dk] C-order (matches the
  //      reference's un-transposed reshape to [b,s,d]).
  _Float16* crow = Ctxh + hbase;
#pragma unroll
  for (int r = 0; r < 8; ++r) {
    const int row = q0 + r + 8 * kh;
    const float inv = 1.f / lrow[r];
#pragma unroll
    for (int t = 0; t < 4; ++t)
      crow[(size_t)row * DK + t * 16 + lane] = (_Float16)(acc[t][r] * inv);
  }
}

// ---------------------------------------------------------------------------
// Kernel 3: output projection + bias + residual (f16 GEMM, f32 epilogue).
// ---------------------------------------------------------------------------
__global__ void oproj_kernel(const _Float16* __restrict__ ctxh,
                             const _Float16* __restrict__ Who,
                             const float* __restrict__ bo,
                             const float* __restrict__ x,
                             float* __restrict__ out) {
  const int l    = threadIdx.x;
  const int lane = l & 15, kh = l >> 4;
  const int n0 = blockIdx.x * 16;
  const int m0 = blockIdx.y * 16;

  const _Float16* crow = ctxh + (size_t)(m0 + lane) * D_MODEL;
  const _Float16* wrow = Who  + (size_t)(n0 + lane) * D_MODEL;

  v8f acc = {};
#pragma unroll
  for (int k0 = 0; k0 < D_MODEL; k0 += 32) {
    v16h a = loadA_h(crow, k0, kh);
    v16h b = loadB16_h(wrow + k0 + kh * 16);
    acc = wmma16(a, b, acc);
  }

  const float bn = bo[n0 + lane];
#pragma unroll
  for (int r = 0; r < 8; ++r) {
    const int M = m0 + r + 8 * kh;
    const size_t idx = (size_t)M * D_MODEL + n0 + lane;
    out[idx] = acc[r] + bn + x[idx];
  }
}

// ---------------------------------------------------------------------------
// Launch: cvt(x,W*) -> qkv -> flash attention -> output projection.
// Workspace (halfs): xh 4Mi | Wh 4x256Ki | Qh 4Mi | Kh 4Mi | VhT 4Mi | Ctx 4Mi
// = 22,020,096 halfs = 42 MiB.
// ---------------------------------------------------------------------------
extern "C" void kernel_launch(void* const* d_in, const int* in_sizes, int n_in,
                              void* d_out, int out_size, void* d_ws, size_t ws_size,
                              hipStream_t stream) {
  (void)in_sizes; (void)n_in; (void)out_size; (void)ws_size;

  const float* x    = (const float*)d_in[0];
  const int*   mask = (const int*)  d_in[1];
  const float* Wq   = (const float*)d_in[2];
  const float* bq   = (const float*)d_in[3];
  const float* Wk   = (const float*)d_in[4];
  const float* bk   = (const float*)d_in[5];
  const float* Wv   = (const float*)d_in[6];
  const float* bv   = (const float*)d_in[7];
  const float* Wo   = (const float*)d_in[8];
  const float* bo   = (const float*)d_in[9];
  float*       out  = (float*)d_out;

  const size_t P  = (size_t)BATCH * SEQ * D_MODEL;   // 4 Mi elements
  const size_t WE = (size_t)D_MODEL * D_MODEL;       // 256 Ki elements
  _Float16* xh   = (_Float16*)d_ws;
  _Float16* Whq  = xh + P;
  _Float16* Whk  = Whq + WE;
  _Float16* Whv  = Whk + WE;
  _Float16* Who  = Whv + WE;
  _Float16* Qh   = Who + WE;
  _Float16* Kh   = Qh + P;
  _Float16* VhT  = Kh + P;
  _Float16* Ctxh = VhT + P;

  // ---- precision prep: one-time f32 -> f16 of streamed operands
  cvt_f32_f16<<<dim3((P / 4 + 255) / 256), dim3(256), 0, stream>>>(x, xh, (int)(P / 4));
  cvt_f32_f16<<<dim3((WE / 4 + 255) / 256), dim3(256), 0, stream>>>(Wq, Whq, (int)(WE / 4));
  cvt_f32_f16<<<dim3((WE / 4 + 255) / 256), dim3(256), 0, stream>>>(Wk, Whk, (int)(WE / 4));
  cvt_f32_f16<<<dim3((WE / 4 + 255) / 256), dim3(256), 0, stream>>>(Wv, Whv, (int)(WE / 4));
  cvt_f32_f16<<<dim3((WE / 4 + 255) / 256), dim3(256), 0, stream>>>(Wo, Who, (int)(WE / 4));

  dim3 blk(32, 1, 1);                                // one wave32
  dim3 g1(D_MODEL / 16, (BATCH * SEQ) / 16, 3);
  qkv_kernel<<<g1, blk, 0, stream>>>(xh, Whq, Whk, Whv, bq, bk, bv, Qh, Kh, VhT);

  dim3 g2(SEQ / 16, NHEAD, BATCH);
  attn_kernel<<<g2, blk, 0, stream>>>(Qh, Kh, VhT, mask, Ctxh);

  dim3 g3(D_MODEL / 16, (BATCH * SEQ) / 16, 1);
  oproj_kernel<<<g3, blk, 0, stream>>>(Ctxh, Who, bo, x, out);
}
